// SMPModel_13159779795349
// MI455X (gfx1250) — matrix operand order repaired and checked by reading, not compiled, not run on hardware
//
#include <hip/hip_runtime.h>
#include <math.h>

typedef __attribute__((ext_vector_type(16))) _Float16 v16h;
typedef __attribute__((ext_vector_type(8)))  _Float16 v8h;
typedef __attribute__((ext_vector_type(8)))  float    v8f;

#define ALPHA   0.05f
#define SIGMA   0.2f
#define THETA   0.3f
#define A_PAR   0.02f
#define BEFF    0.1f      /* sign(RHO)*B = +0.1 */
#define MULT    2.0f
#define N_STEPS 100
#define DT      0.01f
#define HID     64
#define BATCH   16384
#define PEN_SC  100.0f
#define LOWERB  1e-6f
#define CR  (A_PAR - 0.5f*BEFF*BEFF)          /* R drift:  0.015  */
#define CP  (-(ALPHA + 0.5f*THETA*THETA))     /* p drift: -0.095  */
#define LOG2E 1.4426950408889634f

union Av { v16h v; _Float16 e[16]; v8h h[2]; };
union Cv { v8f  v; float    e[8]; };

static __device__ inline v8f wmma16(v16h a, v16h b, v8f c) {
  return __builtin_amdgcn_wmma_f32_16x16x32_f16(false, a, false, b, (short)0, c, false, false);
}

/* branch-free exp: args here are |x| < ~2, no range reduction needed */
static __device__ inline float fast_exp(float x) {
#if __has_builtin(__builtin_amdgcn_exp2f)
  return __builtin_amdgcn_exp2f(x * LOG2E);
#else
  return __expf(x);
#endif
}

/* hardware tanh on gfx1250 (v_tanh_f32); branch-free fallback */
static __device__ inline float fast_tanh(float x) {
#if __has_builtin(__builtin_amdgcn_tanhf)
  return __builtin_amdgcn_tanhf(x);
#else
  float xc = fminf(fmaxf(x, -9.f), 9.f);
#if __has_builtin(__builtin_amdgcn_exp2f) && __has_builtin(__builtin_amdgcn_rcpf)
  float e = __builtin_amdgcn_exp2f(xc * (2.f * LOG2E));
  return 1.f - 2.f * __builtin_amdgcn_rcpf(e + 1.f);
#else
  float e = __expf(2.f * xc);
  return (e - 1.f) / (e + 1.f);
#endif
#endif
}

/* ---------- kernel A: p0, dp0, pi0 at the (uniform) t=0 state; zero loss acc ---------- */
__global__ __launch_bounds__(64) void init_kernel(
    const float* X0p, const float* R0p,
    const float* pW1, const float* pb1, const float* pW2, const float* pb2,
    const float* pW3, const float* pb3,
    const float* qW1, const float* qb1, const float* qW2, const float* qb2,
    const float* qW3, const float* qb3, float* ws) {
  __shared__ float h1p[HID], g1p[HID], h1q[HID], red[HID];
  const int j = threadIdx.x;
  const float X0 = X0p[0], R0 = R0p[0];
  /* state = [MULT*0, X0, R0]; x is state column 1 */
  {
    float pre = X0 * pW1[HID + j] + R0 * pW1[2 * HID + j] + pb1[j];
    float h = tanhf(pre);
    h1p[j] = h;
    g1p[j] = (1.f - h * h) * pW1[HID + j];
    float preq = X0 * qW1[HID + j] + R0 * qW1[2 * HID + j] + qb1[j];
    h1q[j] = tanhf(preq);
  }
  __syncthreads();
  float ap = 0.f, ag = 0.f, aq = 0.f;
  for (int k = 0; k < HID; ++k) {
    ap += h1p[k] * pW2[k * HID + j];
    ag += g1p[k] * pW2[k * HID + j];
    aq += h1q[k] * qW2[k * HID + j];
  }
  float h2  = tanhf(ap + pb2[j]);
  float g2  = (1.f - h2 * h2) * ag;
  float h2q = tanhf(aq + qb2[j]);
  float tp = h2 * pW3[j], td = g2 * pW3[j], tq = h2q * qW3[j];

  auto reduce = [&](float v) -> float {
    red[j] = v; __syncthreads();
    for (int s = 32; s > 0; s >>= 1) { if (j < s) red[j] += red[j + s]; __syncthreads(); }
    float r = red[0]; __syncthreads(); return r;
  };
  float p0  = reduce(tp) + pb3[0];
  float dp0 = reduce(td);
  float pi0 = reduce(tq) + qb3[0];
  if (j == 0) { ws[0] = p0; ws[1] = dp0; ws[2] = pi0; ws[3] = 0.f; ws[4] = 0.f; }
}

/* ---------- kernel B: main time-march, WMMA MLP per step ---------- */
__global__ __launch_bounds__(128) void bsde_kernel(
    const float* __restrict__ dw, const float* X0p, const float* R0p,
    const float* __restrict__ qW1, const float* __restrict__ qb1,
    const float* __restrict__ qW2, const float* __restrict__ qb2,
    const float* __restrict__ qW3, const float* __restrict__ qb3,
    const float* __restrict__ ws, float* __restrict__ out,
    float* __restrict__ lossAcc) {
  __shared__ __align__(32) _Float16 sW1[3][HID];
  __shared__ __align__(32) _Float16 sW2[HID][HID];
  __shared__ __align__(32) _Float16 sW3[HID];
  __shared__ float sb1[HID], sb2[HID];
  __shared__ __align__(32) _Float16 Hbuf[4][32][72];   /* padded rows: 144B stride */
  __shared__ __align__(32) float piBuf[4][32];

  const int tid = threadIdx.x;
  for (int i = tid; i < 3 * HID; i += 128) ((_Float16*)sW1)[i] = (_Float16)qW1[i];
  for (int i = tid; i < HID * HID; i += 128) ((_Float16*)sW2)[i] = (_Float16)qW2[i];
  if (tid < HID) { sW3[tid] = (_Float16)qW3[tid]; sb1[tid] = qb1[tid]; sb2[tid] = qb2[tid]; }
  __syncthreads();

  const int w = tid >> 5, lane = tid & 31;
  const int m = lane & 15, hi = lane >> 4;
  const int gs = (blockIdx.x * 4 + w) * 32 + lane;

  /* --- persistent B fragments (lane%16 = N col, slot = K in table order) --- */
  v16h bW1f[4], bW2f[2][4], bW3f[2];
#pragma unroll
  for (int nt = 0; nt < 4; ++nt) {
    Av a;
#pragma unroll
    for (int q = 0; q < 16; ++q) a.e[q] = (_Float16)0.f;
    if (hi == 0) { for (int q = 0; q < 3; ++q) a.e[q] = sW1[q][nt * 16 + m]; }
    bW1f[nt] = a.v;
  }
#pragma unroll
  for (int kc = 0; kc < 2; ++kc)
#pragma unroll
    for (int nt = 0; nt < 4; ++nt) {
      Av a;
#pragma unroll
      for (int s = 0; s < 16; ++s)
        a.e[s] = sW2[kc * 32 + (s < 8 ? 8 * hi + s : 8 + 8 * hi + s)][nt * 16 + m];
      bW2f[kc][nt] = a.v;
    }
#pragma unroll
  for (int kc = 0; kc < 2; ++kc) {
    Av a;
#pragma unroll
    for (int q = 0; q < 16; ++q) a.e[q] = (_Float16)0.f;
    if (m == 0) { for (int s = 0; s < 16; ++s) a.e[s] = sW3[kc * 32 + (s < 8 ? 8 * hi + s : 8 + 8 * hi + s)]; }
    bW3f[kc] = a.v;
  }
  /* hoist per-lane bias values out of the time loop */
  float b1v[4], b2v[4];
#pragma unroll
  for (int nt = 0; nt < 4; ++nt) { b1v[nt] = sb1[nt * 16 + m]; b2v[nt] = sb2[nt * 16 + m]; }

  const float X0 = X0p[0], R0 = R0p[0];
  const float p0 = ws[0], mdp0 = -ws[1], pi0 = ws[2];
  const float b3v = qb3[0];

  /* t = 0 row */
  { float* o0 = out + (size_t)gs * 5;
    o0[0] = X0; o0[1] = R0; o0[2] = pi0; o0[3] = -p0; o0[4] = mdp0; }

  float x = X0, minx = X0, Wsum = 0.f, piC = pi0, Rv = R0, p = p0;

#pragma unroll 1
  for (int i = 0; i < N_STEPS; ++i) {
    float dwi = dw[(size_t)gs * N_STEPS + i];
    if (i + 4 < N_STEPS) __builtin_prefetch(&dw[(size_t)gs * N_STEPS + i + 4], 0, 1);
    x = x + x * (ALPHA * DT) + piC * (SIGMA * THETA * DT) + piC * SIGMA * dwi;
    minx = fminf(minx, x);
    Wsum += dwi;
    const float tn = (float)(i + 1) * DT;
    Rv = R0 * fast_exp(CR * tn + BEFF * Wsum);
    p  = p0 * fast_exp(CP * tn - THETA * Wsum);

    if (i < N_STEPS - 1) {
      const float t2 = MULT * tn;
      /* ---- layer 1: S(32x3 pad 32) @ W1(3x64) ---- */
#pragma unroll
      for (int ma = 0; ma < 2; ++ma) {
        float xs = __shfl(x,  ma * 16 + m, 32);
        float Rs = __shfl(Rv, ma * 16 + m, 32);
        Av a;
#pragma unroll
        for (int q = 0; q < 16; ++q) a.e[q] = (_Float16)0.f;
        if (hi == 0) { a.e[0] = (_Float16)t2; a.e[1] = (_Float16)xs; a.e[2] = (_Float16)Rs; }
#pragma unroll
        for (int nt = 0; nt < 4; ++nt) {
          Cv c;
#pragma unroll
          for (int r = 0; r < 8; ++r) c.e[r] = b1v[nt];
          c.v = wmma16(a.v, bW1f[nt], c.v);
#pragma unroll
          for (int r = 0; r < 8; ++r)
            Hbuf[w][ma * 16 + r + 8 * hi][nt * 16 + m] = (_Float16)fast_tanh(c.e[r]);
        }
      }
      /* ---- layer 2: H1(32x64) @ W2(64x64) ---- */
#pragma unroll
      for (int ma = 0; ma < 2; ++ma) {
        const int row = ma * 16 + m;
        Av a0, a1;
        a0.h[0] = *(const v8h*)&Hbuf[w][row][8 * hi];
        a0.h[1] = *(const v8h*)&Hbuf[w][row][8 * hi + 16];
        a1.h[0] = *(const v8h*)&Hbuf[w][row][32 + 8 * hi];
        a1.h[1] = *(const v8h*)&Hbuf[w][row][32 + 8 * hi + 16];
#pragma unroll
        for (int nt = 0; nt < 4; ++nt) {
          Cv c;
#pragma unroll
          for (int r = 0; r < 8; ++r) c.e[r] = b2v[nt];
          c.v = wmma16(a0.v, bW2f[0][nt], c.v);
          c.v = wmma16(a1.v, bW2f[1][nt], c.v);
#pragma unroll
          for (int r = 0; r < 8; ++r)
            Hbuf[w][ma * 16 + r + 8 * hi][nt * 16 + m] = (_Float16)fast_tanh(c.e[r]);
        }
      }
      /* ---- layer 3: H2(32x64) @ W3(64x1 pad 16) ---- */
#pragma unroll
      for (int ma = 0; ma < 2; ++ma) {
        const int row = ma * 16 + m;
        Av a0, a1;
        a0.h[0] = *(const v8h*)&Hbuf[w][row][8 * hi];
        a0.h[1] = *(const v8h*)&Hbuf[w][row][8 * hi + 16];
        a1.h[0] = *(const v8h*)&Hbuf[w][row][32 + 8 * hi];
        a1.h[1] = *(const v8h*)&Hbuf[w][row][32 + 8 * hi + 16];
        Cv c;
#pragma unroll
        for (int r = 0; r < 8; ++r) c.e[r] = b3v;
        c.v = wmma16(a0.v, bW3f[0], c.v);
        c.v = wmma16(a1.v, bW3f[1], c.v);
        if (m == 0) *(v8f*)&piBuf[w][ma * 16 + 8 * hi] = c.v;  /* column N=0 */
      }
      piC = piBuf[w][lane];
    }

    float* orow = out + ((size_t)(i + 1) * BATCH + gs) * 5;
    orow[0] = x; orow[1] = Rv; orow[2] = piC; orow[3] = -p; orow[4] = mdp0;
  }

  /* ---- losses ---- */
  const float xc = fmaxf(x, LOWERB);
  const float sq = sqrtf(xc);
  const float ux = Rv / sq;            /* R_fin * xc^(gamma-1), gamma=0.5 */
  const float uv = 2.f * Rv * sq;      /* R_fin * xc^gamma / gamma        */
  const float d  = fmaxf(LOWERB - minx, 0.f);
  const float pen = PEN_SC * d * d;
  float t1 = (p + ux) * (p + ux) + pen;
  float t2 = -uv + pen;
#pragma unroll
  for (int off = 16; off > 0; off >>= 1) {
    t1 += __shfl_xor(t1, off, 32);
    t2 += __shfl_xor(t2, off, 32);
  }
  if (lane == 0) { atomicAdd(&lossAcc[0], t1); atomicAdd(&lossAcc[1], t2); }
}

/* ---------- kernel C: finalize losses ---------- */
__global__ void fin_kernel(const float* ws, float* out) {
  const float inv = 1.0f / (float)BATCH;
  const float lp = ws[3] * inv;
  const float lq = ws[4] * inv;
  const size_t base = (size_t)(N_STEPS + 1) * BATCH * 5;
  out[base + 0] = lp;
  out[base + 1] = lp + lq;
}

extern "C" void kernel_launch(void* const* d_in, const int* in_sizes, int n_in,
                              void* d_out, int out_size, void* d_ws, size_t ws_size,
                              hipStream_t stream) {
  const float* dw  = (const float*)d_in[0];
  const float* X0  = (const float*)d_in[1];
  const float* R0  = (const float*)d_in[2];
  const float* pW1 = (const float*)d_in[3];
  const float* pb1 = (const float*)d_in[4];
  const float* pW2 = (const float*)d_in[5];
  const float* pb2 = (const float*)d_in[6];
  const float* pW3 = (const float*)d_in[7];
  const float* pb3 = (const float*)d_in[8];
  const float* qW1 = (const float*)d_in[9];
  const float* qb1 = (const float*)d_in[10];
  const float* qW2 = (const float*)d_in[11];
  const float* qb2 = (const float*)d_in[12];
  const float* qW3 = (const float*)d_in[13];
  const float* qb3 = (const float*)d_in[14];
  float* out = (float*)d_out;
  float* ws  = (float*)d_ws;

  init_kernel<<<1, 64, 0, stream>>>(X0, R0, pW1, pb1, pW2, pb2, pW3, pb3,
                                    qW1, qb1, qW2, qb2, qW3, qb3, ws);
  bsde_kernel<<<BATCH / 128, 128, 0, stream>>>(dw, X0, R0, qW1, qb1, qW2, qb2,
                                               qW3, qb3, ws, out, ws + 3);
  fin_kernel<<<1, 1, 0, stream>>>(ws, out);
}
